// LocalSpatialAttention_58119497450441
// MI455X (gfx1250) — matrix operand order repaired and checked by reading, not compiled, not run on hardware
//
#include <hip/hip_runtime.h>
#include <hip/hip_bf16.h>

typedef __attribute__((ext_vector_type(16))) _Float16 v16h;
typedef __attribute__((ext_vector_type(8)))  float    v8f;

#define Bc 4
#define Yc 192
#define Xc 192
#define Cc 128
#define HFc 128
#define NPIX ((size_t)Bc * Yc * Xc)   // 147456
#define PW 5

__device__ __forceinline__ int clampi(int v, int lo, int hi) {
    return v < lo ? lo : (v > hi ? hi : v);
}

// ---------------------------------------------------------------------------
// Kernel 0: pack fp32 weights [128,128] (K-major rows, N contiguous) into the
// f16 WMMA B order used here (K order as A): [j=0..7 N-tiles][c=0..3 K-chunks][lane 0..31][16]
//   lane<16 : column N = 16j+lane, elements = K (32c+0 .. 32c+15)
//   lane>=16: column N = 16j+lane-16, elements = K (32c+16 .. 32c+31)
// ---------------------------------------------------------------------------
__global__ void pack_weights(const float* __restrict__ Wq,
                             const float* __restrict__ Wk,
                             const float* __restrict__ Wv,
                             const float* __restrict__ Wo,
                             _Float16* __restrict__ out)   // 4 * 8*4*32*16 halves
{
    int tid = blockIdx.x * blockDim.x + threadIdx.x;       // 0..4095
    if (tid >= 4 * 1024) return;
    int mat  = tid >> 10;
    int r    = tid & 1023;                                  // (j*4+c)*32 + lane
    int lane = r & 31;
    int jc   = r >> 5;
    int c    = jc & 3;
    int j    = jc >> 2;
    const float* W = (mat == 0) ? Wq : (mat == 1) ? Wk : (mat == 2) ? Wv : Wo;
    int n     = j * 16 + (lane & 15);
    int kbase = c * 32 + (lane >> 4) * 16;
    _Float16* dst = out + (size_t)tid * 16;
#pragma unroll
    for (int i = 0; i < 16; ++i)
        dst[i] = (_Float16)W[(size_t)(kbase + i) * HFc + n];
}

// ---------------------------------------------------------------------------
// Kernel 1/3: C[M,128] = f16wmma( A[M,128] , Wpack ) (+ optional query pos-emb)
// One block = 4 waves, one 16-row tile; wave w covers columns 32w..32w+31.
// K = 128 = 4 x (16x16x32 f16 WMMA) per N-tile, fp32 accumulation.
// ---------------------------------------------------------------------------
__global__ __launch_bounds__(128)
void gemm_proj(const float* __restrict__ A,          // [M,128] fp32
               const _Float16* __restrict__ Wpack,   // [8][4][32][16] f16
               float* __restrict__ Out,              // [M,128] fp32
               const float* __restrict__ pos_emb)    // [25,128] fp32 or nullptr
{
    const int wave = threadIdx.x >> 5;
    const int lane = threadIdx.x & 31;
    const int m0   = blockIdx.x * 16;

    const int mrow  = m0 + (lane & 15);
    const int khalf = (lane >> 4) * 16;  // A k order used here (matches Wpack): lane<16 -> K 0-15, lane>=16 -> K 16-31

    v8f acc[2] = { v8f{}, v8f{} };

#pragma unroll
    for (int c = 0; c < 4; ++c) {
        const float* ap = A + (size_t)mrow * Cc + c * 32 + khalf;
        float4 a0 = *(const float4*)(ap);
        float4 a1 = *(const float4*)(ap + 4);
        float4 a2 = *(const float4*)(ap + 8);
        float4 a3 = *(const float4*)(ap + 12);
        v16h a;
        a[0]=(_Float16)a0.x;  a[1]=(_Float16)a0.y;  a[2]=(_Float16)a0.z;  a[3]=(_Float16)a0.w;
        a[4]=(_Float16)a1.x;  a[5]=(_Float16)a1.y;  a[6]=(_Float16)a1.z;  a[7]=(_Float16)a1.w;
        a[8]=(_Float16)a2.x;  a[9]=(_Float16)a2.y;  a[10]=(_Float16)a2.z; a[11]=(_Float16)a2.w;
        a[12]=(_Float16)a3.x; a[13]=(_Float16)a3.y; a[14]=(_Float16)a3.z; a[15]=(_Float16)a3.w;

#pragma unroll
        for (int jj = 0; jj < 2; ++jj) {
            int j = wave * 2 + jj;
            v16h b = *(const v16h*)(Wpack + ((size_t)(j * 4 + c) * 32 + lane) * 16);
            acc[jj] = __builtin_amdgcn_wmma_f32_16x16x32_f16(
                false, a, false, b, (short)0, acc[jj], false, false);
        }
    }

    // C/D layout: element r of lane l -> M = r + 8*(l>>4), N = (l&15)
    const int colbase = lane & 15;
    const int rowoff  = (lane >> 4) * 8;
#pragma unroll
    for (int jj = 0; jj < 2; ++jj) {
        int col = wave * 32 + jj * 16 + colbase;
#pragma unroll
        for (int r = 0; r < 8; ++r) {
            int row = m0 + rowoff + r;
            float v = acc[jj][r];
            if (pos_emb != nullptr) {
                int xy   = row % (Yc * Xc);
                int yy   = xy / Xc;
                int xx   = xy - yy * Xc;
                int yrel = yy - clampi(yy, 2, Yc - 3) + 2;
                int xrel = xx - clampi(xx, 2, Xc - 3) + 2;
                v += pos_emb[(size_t)(yrel * PW + xrel) * HFc + col];
            }
            Out[(size_t)row * HFc + col] = v;
        }
    }
}

// ---------------------------------------------------------------------------
// Kernel 2: local 5x5 attention. Block = 4x4 pixel tile x 8 heads (128 thr).
// K/V 8x8 halo + pos_emb staged in LDS. The K/V LDS layout is bank-skewed:
//   float4 index = pos*33 + h*4 + fchunk
// so a wave's 32 (pos,h) lanes cover all 16 bank-groups exactly twice ->
// optimal 2-pass ds_load_b128 / ds_store_b128 (vs 8-way conflicts unskewed).
// Online softmax in registers. Output overwrites Q in place.
// ---------------------------------------------------------------------------
#define TS 4
#define POSSTR 33   // float4 stride per halo position (32 data + 1 skew)
__global__ __launch_bounds__(128)
void local_attn(float* __restrict__ Q,          // [B,Y,X,128] in: Q+pos, out: attn
                const float* __restrict__ K,
                const float* __restrict__ V,
                const float* __restrict__ pos_emb)  // [25,128]
{
    __shared__ float4 sK4[64 * POSSTR];         // 8x8 halo, skewed
    __shared__ float4 sV4[64 * POSSTR];
    __shared__ float4 sP4[25 * 32];             // pos_emb, f4 idx = s*32 + h*4 + fc

    const int b  = blockIdx.z;
    const int ty = blockIdx.y * TS;
    const int tx = blockIdx.x * TS;
    const int ybase = clampi(ty, 2, Yc - 3) - 2;
    const int xbase = clampi(tx, 2, Xc - 3) - 2;
    const int tid = threadIdx.x;

    // stage 8x8x128 halo of K and V. Per wave-instruction: one pos, c4 = lane,
    // dst bank-group = (pos + lane) mod 16 -> optimal 2-way stores.
#pragma unroll
    for (int it = 0; it < 16; ++it) {
        int q4  = it * 128 + tid;        // global float4 index, 2048 total
        int c4  = q4 & 31;               // 32 float4 per position
        int pos = q4 >> 5;
        int hx = pos & 7, hy = pos >> 3;
        int gy = min(ybase + hy, Yc - 1);
        int gx = min(xbase + hx, Xc - 1);
        size_t src4 = (((size_t)(b * Yc + gy) * Xc + gx) * HFc) / 4 + c4;
        sK4[pos * POSSTR + c4] = ((const float4*)K)[src4];
        sV4[pos * POSSTR + c4] = ((const float4*)V)[src4];
    }
    for (int q4 = tid; q4 < 25 * 32; q4 += 128)
        sP4[q4] = ((const float4*)pos_emb)[q4];
    __syncthreads();

    // thread -> (pixel, head): h fastest so Q loads are contiguous across the block
    const int h  = tid & 7;
    const int p  = tid >> 3;
    const int px = p & 3, py = p >> 2;
    const int y = ty + py, x = tx + px;
    const int cy = clampi(y, 2, Yc - 3);
    const int cx = clampi(x, 2, Xc - 3);

    const size_t qoff4 = ((((size_t)(b * Yc + y) * Xc + x) * HFc) + h * 16) / 4;
    float4 q[4];
#pragma unroll
    for (int fc = 0; fc < 4; ++fc) q[fc] = ((const float4*)Q)[qoff4 + fc];

    const float scale = 0.25f;           // 1/sqrt(F=16)
    float m = -1e9f, sum = 0.f;
    float4 acc[4];
#pragma unroll
    for (int fc = 0; fc < 4; ++fc) acc[fc] = make_float4(0.f, 0.f, 0.f, 0.f);

#pragma unroll
    for (int dy = -2; dy <= 2; ++dy) {
#pragma unroll
        for (int dx = -2; dx <= 2; ++dx) {
            const int pos = (cy + dy - ybase) * 8 + (cx + dx - xbase);
            const float4* k4  = sK4 + pos * POSSTR + h * 4;
            const float4* v4  = sV4 + pos * POSSTR + h * 4;
            const float4* pe4 = sP4 + ((dy + 2) * PW + (dx + 2)) * 32 + h * 4;

            float4 kk[4];
            float s = 0.f;
#pragma unroll
            for (int fc = 0; fc < 4; ++fc) {
                kk[fc] = k4[fc];
                float4 pp = pe4[fc];
                s += q[fc].x * (kk[fc].x + pp.x);
                s += q[fc].y * (kk[fc].y + pp.y);
                s += q[fc].z * (kk[fc].z + pp.z);
                s += q[fc].w * (kk[fc].w + pp.w);
            }
            s *= scale;
            float nm   = fmaxf(m, s);
            float corr = __expf(m - nm);
            float pw   = __expf(s - nm);
            sum = sum * corr + pw;
#pragma unroll
            for (int fc = 0; fc < 4; ++fc) {
                float4 vv = v4[fc];
                acc[fc].x = acc[fc].x * corr + pw * vv.x;
                acc[fc].y = acc[fc].y * corr + pw * vv.y;
                acc[fc].z = acc[fc].z * corr + pw * vv.z;
                acc[fc].w = acc[fc].w * corr + pw * vv.w;
            }
            m = nm;
        }
    }

    const float inv = 1.f / (sum + 1e-8f);
#pragma unroll
    for (int fc = 0; fc < 4; ++fc) {
        float4 o = make_float4(acc[fc].x * inv, acc[fc].y * inv,
                               acc[fc].z * inv, acc[fc].w * inv);
        ((float4*)Q)[qoff4 + fc] = o;
    }
}

// ---------------------------------------------------------------------------
extern "C" void kernel_launch(void* const* d_in, const int* in_sizes, int n_in,
                              void* d_out, int out_size, void* d_ws, size_t ws_size,
                              hipStream_t stream) {
    const float* x   = (const float*)d_in[0];
    const float* Wq  = (const float*)d_in[1];
    const float* Wk  = (const float*)d_in[2];
    const float* Wv  = (const float*)d_in[3];
    const float* Wo  = (const float*)d_in[4];
    const float* pe  = (const float*)d_in[5];

    const size_t MATF = NPIX * HFc;                 // 18,874,368 floats each
    float* Qb = (float*)d_ws;
    float* Kb = Qb + MATF;
    float* Vb = Kb + MATF;
    _Float16* Wp  = (_Float16*)(Vb + MATF);         // 4 x 16384 halves
    _Float16* WpQ = Wp;
    _Float16* WpK = Wp + 16384;
    _Float16* WpV = Wp + 2 * 16384;
    _Float16* WpO = Wp + 3 * 16384;

    // 0) pack all four weight matrices into the WMMA-B f16 layout
    pack_weights<<<16, 256, 0, stream>>>(Wq, Wk, Wv, Wo, Wp);

    // 1) Q/K/V projections (Q gets the query positional embedding fused in)
    const int mtiles = (int)(NPIX / 16);            // 9216
    gemm_proj<<<mtiles, 128, 0, stream>>>(x, WpQ, Qb, pe);
    gemm_proj<<<mtiles, 128, 0, stream>>>(x, WpK, Kb, nullptr);
    gemm_proj<<<mtiles, 128, 0, stream>>>(x, WpV, Vb, nullptr);

    // 2) local 5x5 attention, writes in place over Qb
    dim3 agrid(Xc / TS, Yc / TS, Bc);               // 48 x 48 x 4
    local_attn<<<agrid, 128, 0, stream>>>(Qb, Kb, Vb, pe);

    // 3) output projection into d_out
    gemm_proj<<<mtiles, 128, 0, stream>>>(Qb, WpO, (float*)d_out, nullptr);
}
